// CrossAttention_78314433675808
// MI455X (gfx1250) — matrix operand order repaired and checked by reading, not compiled, not run on hardware
//
#include <hip/hip_runtime.h>
#include <hip/hip_bf16.h>

typedef __attribute__((ext_vector_type(16))) _Float16 v16h;
typedef __attribute__((ext_vector_type(8)))  _Float16 v8h;
typedef __attribute__((ext_vector_type(8)))  float    v8f;

#define B_    8
#define L_    1024
#define QD_   1024
#define KVD_  768
#define H_    16
#define DH_   64
#define EPS_  1e-5f
#define LOG2E 1.44269504088896340736f

// ---------------------------------------------------------------- helpers

__device__ __forceinline__ v8f vzero8() {
  v8f z;
#pragma unroll
  for (int i = 0; i < 8; ++i) z[i] = 0.0f;
  return z;
}

__device__ __forceinline__ v16h combine8(v8h g0, v8h g1) {
  v16h r;
#pragma unroll
  for (int i = 0; i < 8; ++i) { r[i] = g0[i]; r[8 + i] = g1[i]; }
  return r;
}

// D = A(16x32 f16) * B(32x16 f16) + C(16x16 f32)
__device__ __forceinline__ v8f wmma16x16x32(v16h a, v16h b, v8f c) {
  return __builtin_amdgcn_wmma_f32_16x16x32_f16(
      /*neg_a=*/false, a, /*neg_b=*/false, b,
      /*c_mod=*/(short)0, c, /*reuse_a=*/false, /*reuse_b=*/false);
}

// A-tile (16x32) from row-major f16 [rows][ldk]:
// lane: c=lane>>4, m=lane&15; element e -> K = (e/8)*16 + c*8 + e%8
__device__ __forceinline__ v16h load_a_f16(const _Float16* __restrict__ X, int ldk,
                                           int rowBase, int k0, int lane) {
  const int c = lane >> 4, m = lane & 15;
  const _Float16* p = X + (size_t)(rowBase + m) * ldk + k0 + c * 8;
  return combine8(*reinterpret_cast<const v8h*>(p),
                  *reinterpret_cast<const v8h*>(p + 16));
}

// B-tile (32x16): B(k,n) = W[col+n][k0+k], W row-major f16 [Cout][ldk]
// lane: c=lane>>4, n=lane&15; element e -> K = (e/8)*16 + c*8 + e%8
__device__ __forceinline__ v16h load_b_f16(const _Float16* __restrict__ W, int ldk,
                                           int col, int k0, int lane) {
  const int c = lane >> 4, n = lane & 15;
  const _Float16* p = W + (size_t)(col + n) * ldk + k0 + c * 8;
  return combine8(*reinterpret_cast<const v8h*>(p),
                  *reinterpret_cast<const v8h*>(p + 16));
}

// ---------------------------------------------------------------- kernel 0
// One-shot fp32 -> fp16 conversion, 8 elements per thread (2x b128 in, 1x b128 out)
__global__ __launch_bounds__(256) void cvt_f32_f16_kernel(
    const float* __restrict__ src, _Float16* __restrict__ dst, int n8) {
  const int i = blockIdx.x * 256 + threadIdx.x;
  if (i >= n8) return;
  const float4* s = reinterpret_cast<const float4*>(src) + 2 * (size_t)i;
  const float4 f0 = s[0], f1 = s[1];
  v8h h;
  h[0] = (_Float16)f0.x; h[1] = (_Float16)f0.y; h[2] = (_Float16)f0.z; h[3] = (_Float16)f0.w;
  h[4] = (_Float16)f1.x; h[5] = (_Float16)f1.y; h[6] = (_Float16)f1.z; h[7] = (_Float16)f1.w;
  reinterpret_cast<v8h*>(dst)[i] = h;
}

// ---------------------------------------------------------------- kernel A
// Y[b][h][l][dh] (f16) = X16[b*L+l][:] @ W16.T + bias ; wave: 16 rows x 64 cols
__global__ __launch_bounds__(256) void proj_head_kernel(
    const _Float16* __restrict__ X, const _Float16* __restrict__ W,
    const float* __restrict__ bias, _Float16* __restrict__ Y, int Cin) {
  const int lane = threadIdx.x & 31;
  const int wave = threadIdx.x >> 5;
  const int c = lane >> 4, nlo = lane & 15;
  const int rowBase = (blockIdx.x * 8 + wave) * 16;
  const int colBase = blockIdx.y * 64;

  v8f acc[4];
#pragma unroll
  for (int n = 0; n < 4; ++n) acc[n] = vzero8();

  for (int k0 = 0; k0 < Cin; k0 += 32) {
    v16h a = load_a_f16(X, Cin, rowBase, k0, lane);
#pragma unroll
    for (int n = 0; n < 4; ++n) {
      v16h b = load_b_f16(W, Cin, colBase + n * 16, k0, lane);
      acc[n] = wmma16x16x32(a, b, acc[n]);
    }
  }

#pragma unroll
  for (int n = 0; n < 4; ++n) {
    const int col = colBase + n * 16 + nlo;
    const int h = col >> 6, dh = col & 63;
    const float bb = bias[col];
#pragma unroll
    for (int r = 0; r < 8; ++r) {
      const int row = rowBase + r + 8 * c;       // global row in [0, B*L)
      const int b = row >> 10, l = row & 1023;
      Y[(((size_t)(b * H_ + h)) * L_ + l) * DH_ + dh] = (_Float16)(acc[n][r] + bb);
    }
  }
}

// ---------------------------------------------------------------- kernel B
// Flash attention per (b,h); 8 waves x 16 query rows; key blocks of 32.
__global__ __launch_bounds__(256) void attention_kernel(
    const _Float16* __restrict__ Qh, const _Float16* __restrict__ Kh,
    const _Float16* __restrict__ Vh, const int* __restrict__ mask,
    _Float16* __restrict__ Attn) {
  __shared__ _Float16 sK[32 * 64];        // [key][dh]
  __shared__ _Float16 sVt[64 * 40];       // [dh][key], padded stride 40
  __shared__ _Float16 sP[8 * 16 * 32];    // per-wave P scratch [16][32]

  const int tid = threadIdx.x;
  const int lane = tid & 31, wave = tid >> 5;
  const int c = lane >> 4, nlo = lane & 15;
  const int bh = blockIdx.x;
  const int b = bh >> 4, h = bh & 15;
  const int qBase = blockIdx.y * 128 + wave * 16;

  const _Float16* Qp = Qh + (size_t)bh * L_ * DH_;
  const _Float16* Kp = Kh + (size_t)bh * L_ * DH_;
  const _Float16* Vp = Vh + (size_t)bh * L_ * DH_;

  // Q A-tiles for the two k-steps (dh 0..31, 32..63), kept in registers.
  v16h qa[2];
#pragma unroll
  for (int ks = 0; ks < 2; ++ks) {
    const _Float16* p = Qp + (size_t)(qBase + nlo) * DH_ + ks * 32 + c * 8;
    qa[ks] = combine8(*reinterpret_cast<const v8h*>(p),
                      *reinterpret_cast<const v8h*>(p + 16));
  }

  v8f o[4];
#pragma unroll
  for (int n = 0; n < 4; ++n) o[n] = vzero8();
  float mrow[8], lrow[8];
#pragma unroll
  for (int r = 0; r < 8; ++r) { mrow[r] = -3.0e38f; lrow[r] = 0.0f; }

  const float sscale = 0.125f * LOG2E;  // DH^-0.5 * log2(e)

  for (int j0 = 0; j0 < L_; j0 += 32) {
    __syncthreads();
    {  // cooperative stage: K row-major, V transposed
      const int f = tid * 8;
      const int key = f >> 6, dh0 = f & 63;
      *reinterpret_cast<uint4*>(&sK[f]) =
          *reinterpret_cast<const uint4*>(Kp + (size_t)(j0 + key) * DH_ + dh0);
      v8h vv = *reinterpret_cast<const v8h*>(Vp + (size_t)(j0 + key) * DH_ + dh0);
#pragma unroll
      for (int i = 0; i < 8; ++i) sVt[(dh0 + i) * 40 + key] = vv[i];
    }
    __syncthreads();

    const float mb0 = mask[b * L_ + j0 + nlo]      ? 0.0f : -3.0e38f;
    const float mb1 = mask[b * L_ + j0 + 16 + nlo] ? 0.0f : -3.0e38f;

    // S = Q * K^T  (two 16x16 tiles covering 32 keys)
    float tt[2][8];
#pragma unroll
    for (int nh = 0; nh < 2; ++nh) {
      v8f s = vzero8();
#pragma unroll
      for (int ks = 0; ks < 2; ++ks) {
        const _Float16* p = &sK[(nh * 16 + nlo) * 64 + ks * 32 + c * 8];
        v16h bt = combine8(*reinterpret_cast<const v8h*>(p),
                           *reinterpret_cast<const v8h*>(p + 16));
        s = wmma16x16x32(qa[ks], bt, s);
      }
      const float mb = nh ? mb1 : mb0;
#pragma unroll
      for (int r = 0; r < 8; ++r) tt[nh][r] = s[r] * sscale + mb;
    }

    // online softmax (base 2); rows live within 16-lane halves
    float newm[8], alpha[8];
#pragma unroll
    for (int r = 0; r < 8; ++r) {
      float mx = fmaxf(tt[0][r], tt[1][r]);
#pragma unroll
      for (int off = 8; off >= 1; off >>= 1) mx = fmaxf(mx, __shfl_xor(mx, off));
      newm[r] = fmaxf(mrow[r], mx);
      alpha[r] = exp2f(mrow[r] - newm[r]);
      mrow[r] = newm[r];
    }
#pragma unroll
    for (int r = 0; r < 8; ++r) {
      const float p0 = exp2f(tt[0][r] - newm[r]);
      const float p1 = exp2f(tt[1][r] - newm[r]);
      float rs = p0 + p1;
#pragma unroll
      for (int off = 8; off >= 1; off >>= 1) rs += __shfl_xor(rs, off);
      lrow[r] = lrow[r] * alpha[r] + rs;
      const int M = r + 8 * c;
      sP[(wave * 16 + M) * 32 + nlo]      = (_Float16)p0;
      sP[(wave * 16 + M) * 32 + 16 + nlo] = (_Float16)p1;
    }
#pragma unroll
    for (int n = 0; n < 4; ++n)
#pragma unroll
      for (int r = 0; r < 8; ++r) o[n][r] = o[n][r] * alpha[r];

    // wave-local LDS handoff: CDNA5 split DS counter + wave barrier
    asm volatile("s_wait_dscnt 0" ::: "memory");
    __builtin_amdgcn_wave_barrier();

    // P as A-tile
    const _Float16* pp = &sP[(wave * 16 + nlo) * 32 + c * 8];
    v16h pa = combine8(*reinterpret_cast<const v8h*>(pp),
                       *reinterpret_cast<const v8h*>(pp + 16));

    // O += P * V  (4 dh-tiles of 16)
#pragma unroll
    for (int n = 0; n < 4; ++n) {
      const _Float16* p = &sVt[(n * 16 + nlo) * 40 + c * 8];
      v16h bt = combine8(*reinterpret_cast<const v8h*>(p),
                         *reinterpret_cast<const v8h*>(p + 16));
      o[n] = wmma16x16x32(pa, bt, o[n]);
    }
  }

  // normalize and write head-interleaved f16 output [B][L][QD]
#pragma unroll
  for (int r = 0; r < 8; ++r) {
    const float inv = 1.0f / lrow[r];
    const int lq = qBase + r + 8 * c;
    const size_t rowOff = ((size_t)b * L_ + lq) * QD_ + h * DH_;
#pragma unroll
    for (int n = 0; n < 4; ++n)
      Attn[rowOff + n * 16 + nlo] = (_Float16)(o[n][r] * inv);
  }
}

// ---------------------------------------------------------------- kernel C
// out = LayerNorm(Attn @ Wo16.T + bo + q) ; block owns 16 full rows.
__global__ __launch_bounds__(256) void oproj_ln_kernel(
    const _Float16* __restrict__ Attn, const _Float16* __restrict__ Wo,
    const float* __restrict__ bo, const float* __restrict__ qin,
    const float* __restrict__ gamma, const float* __restrict__ beta,
    float* __restrict__ out) {
  extern __shared__ float sY[];  // [16][1024] = 64 KB dynamic LDS
  const int tid = threadIdx.x, lane = tid & 31, wave = tid >> 5;
  const int c = lane >> 4, nlo = lane & 15;
  const int rowBase = blockIdx.x * 16;

  v8f acc[2][4];
#pragma unroll
  for (int cb = 0; cb < 2; ++cb)
#pragma unroll
    for (int n = 0; n < 4; ++n) acc[cb][n] = vzero8();

  for (int k0 = 0; k0 < QD_; k0 += 32) {
    v16h a = load_a_f16(Attn, QD_, rowBase, k0, lane);
#pragma unroll
    for (int cb = 0; cb < 2; ++cb)
#pragma unroll
      for (int n = 0; n < 4; ++n) {
        v16h b = load_b_f16(Wo, QD_, wave * 128 + cb * 64 + n * 16, k0, lane);
        acc[cb][n] = wmma16x16x32(a, b, acc[cb][n]);
      }
  }

#pragma unroll
  for (int cb = 0; cb < 2; ++cb)
#pragma unroll
    for (int n = 0; n < 4; ++n) {
      const int col = wave * 128 + cb * 64 + n * 16 + nlo;
      const float bb = bo[col];
#pragma unroll
      for (int r = 0; r < 8; ++r) {
        const int M = r + 8 * c;
        const int g = rowBase + M;
        sY[M * QD_ + col] = acc[cb][n][r] + bb + qin[(size_t)g * QD_ + col];
      }
    }
  __syncthreads();

  // LayerNorm: each wave handles two rows
#pragma unroll
  for (int t = 0; t < 2; ++t) {
    const int rr = wave * 2 + t;
    float s = 0.0f, sq = 0.0f;
    for (int j = lane; j < QD_; j += 32) {
      const float v = sY[rr * QD_ + j];
      s += v; sq += v * v;
    }
#pragma unroll
    for (int off = 16; off >= 1; off >>= 1) {
      s += __shfl_xor(s, off);
      sq += __shfl_xor(sq, off);
    }
    const float mean = s * (1.0f / QD_);
    const float var = sq * (1.0f / QD_) - mean * mean;
    const float rstd = rsqrtf(var + EPS_);
    const int g = rowBase + rr;
    for (int j = lane; j < QD_; j += 32) {
      const float v = (sY[rr * QD_ + j] - mean) * rstd;
      out[(size_t)g * QD_ + j] = v * gamma[j] + beta[j];
    }
  }
}

// ---------------------------------------------------------------- launch

extern "C" void kernel_launch(void* const* d_in, const int* in_sizes, int n_in,
                              void* d_out, int out_size, void* d_ws, size_t ws_size,
                              hipStream_t stream) {
  (void)in_sizes; (void)n_in; (void)out_size; (void)ws_size;
  const float* q     = (const float*)d_in[0];
  const float* k     = (const float*)d_in[1];
  const float* v     = (const float*)d_in[2];
  const int*   mask  = (const int*)d_in[3];
  const float* Wq    = (const float*)d_in[4];
  const float* bq    = (const float*)d_in[5];
  const float* Wk    = (const float*)d_in[6];
  const float* bk    = (const float*)d_in[7];
  const float* Wv    = (const float*)d_in[8];
  const float* bv    = (const float*)d_in[9];
  const float* Wo    = (const float*)d_in[10];
  const float* bo    = (const float*)d_in[11];
  const float* gamma = (const float*)d_in[12];
  const float* beta  = (const float*)d_in[13];
  float* out = (float*)d_out;

  const size_t MB = (size_t)1 << 20;
  char* ws = (char*)d_ws;
  // head-major f16 projections (live: proj -> attention)
  _Float16* Qh  = (_Float16*)(ws);             // 16 MB
  _Float16* Kh  = (_Float16*)(ws + 16 * MB);   // 16 MB
  _Float16* Vh  = (_Float16*)(ws + 32 * MB);   // 16 MB
  // f16 activation copies (live: cvt -> proj); At reuses q16's region after
  _Float16* q16 = (_Float16*)(ws + 48 * MB);   // 16 MB
  _Float16* k16 = (_Float16*)(ws + 64 * MB);   // 12 MB
  _Float16* v16 = (_Float16*)(ws + 76 * MB);   // 12 MB
  _Float16* At  = (_Float16*)(ws + 48 * MB);   // 16 MB (aliases q16, stream-ordered)
  // f16 weights (live: cvt -> end)
  _Float16* Wq16 = (_Float16*)(ws + 88 * MB);            // 2 MB
  _Float16* Wk16 = (_Float16*)(ws + 90 * MB);            // 1.5 MB
  _Float16* Wv16 = (_Float16*)(ws + 91 * MB + MB / 2);   // 1.5 MB
  _Float16* Wo16 = (_Float16*)(ws + 93 * MB);            // 2 MB

  dim3 blk(256);
  // one-shot f32->f16 conversions (8 elems/thread)
  auto cvt = [&](const float* s, _Float16* d, int n) {
    const int n8 = n / 8;
    cvt_f32_f16_kernel<<<dim3((n8 + 255) / 256), blk, 0, stream>>>(s, d, n8);
  };
  cvt(q,  q16,  B_ * L_ * QD_);
  cvt(k,  k16,  B_ * L_ * KVD_);
  cvt(v,  v16,  B_ * L_ * KVD_);
  cvt(Wq, Wq16, QD_ * QD_);
  cvt(Wk, Wk16, QD_ * KVD_);
  cvt(Wv, Wv16, QD_ * KVD_);
  cvt(Wo, Wo16, QD_ * QD_);

  proj_head_kernel<<<dim3(64, 16), blk, 0, stream>>>(q16, Wq16, bq, Qh, QD_);
  proj_head_kernel<<<dim3(64, 16), blk, 0, stream>>>(k16, Wk16, bk, Kh, KVD_);
  proj_head_kernel<<<dim3(64, 16), blk, 0, stream>>>(v16, Wv16, bv, Vh, KVD_);
  attention_kernel<<<dim3(128, 8), blk, 0, stream>>>(Qh, Kh, Vh, mask, At);
  oproj_ln_kernel<<<dim3(512), blk, 65536, stream>>>(At, Wo16, bo, q, gamma, beta, out);
}
